// DeltaLag_70600672411718
// MI455X (gfx1250) — matrix operand order repaired and checked by reading, not compiled, not run on hardware
//
#include <hip/hip_runtime.h>
#include <hip/hip_bf16.h>
#include <math.h>

typedef __attribute__((ext_vector_type(16))) _Float16 v16h;
typedef __attribute__((ext_vector_type(8)))  _Float16 v8h;
typedef __attribute__((ext_vector_type(8)))  float    v8f;

#define S_TOT 3000
#define F_IN  6
#define NH    128
#define LSEQ  40
#define LMAX  10
#define TOPK  5
#define KPAD  160      // 6 (x) + 128 (h) padded to multiple of 32
#define G4    512      // 4*NH gates
#define NKEY  (S_TOT*LMAX)   // 30000
#define KTILES (NKEY/16)     // 1875
#define TBLK  ((S_TOT + 15)/16) // 188

__device__ __forceinline__ float sigmf(float x){ return 1.0f/(1.0f + expf(-x)); }

// A-matrix fragment (16x32 f16): lane m=lane&15 holds row m.
// halves[0..7]  = A[m][k0 + hi*8 .. +7], halves[8..15] = A[m][k0+16+hi*8 .. +7]
__device__ __forceinline__ v16h ldfragA(const _Float16* row, int k0, int hi){
  v8h a = *(const v8h*)(row + k0 + hi*8);
  v8h b = *(const v8h*)(row + k0 + 16 + hi*8);
  v16h r;
#pragma unroll
  for (int i=0;i<8;i++){ r[i]=a[i]; r[i+8]=b[i]; }
  return r;
}
// B-matrix fragment (32x16 f16): lane n=lane&15 holds column n (stored as a
// contiguous "column row" of K values). halves[0..7] = Bcol[n][k0 + hi*8 .. +7],
// halves[8..15] = Bcol[n][k0+16+hi*8 .. +7]
__device__ __forceinline__ v16h ldfragB(const _Float16* colrow, int k0, int hi){
  v8h a = *(const v8h*)(colrow + k0 + hi*8);
  v8h b = *(const v8h*)(colrow + k0 + 16 + hi*8);
  v16h r;
#pragma unroll
  for (int i=0;i<8;i++){ r[i]=a[i]; r[i+8]=b[i]; }
  return r;
}
__device__ __forceinline__ v8f wmma16(v16h a, v16h b, v8f c){
  return __builtin_amdgcn_wmma_f32_16x16x32_f16(false, a, false, b, (short)0, c, false, false);
}

// ---------------------------------------------------------------------------
// Prep: pack [Wih | Whh | 0] into f16 per-gate rows of length KPAD, and
// convert WQ/WK to f16 row-major (these rows serve as B "column rows").
// ---------------------------------------------------------------------------
__global__ void prep_kernel(const float* __restrict__ Wih,   // [512][6]
                            const float* __restrict__ Whh,   // [512][128]
                            const float* __restrict__ WQ,    // [128][128]
                            const float* __restrict__ WK,    // [128][128]
                            _Float16* __restrict__ Wc,       // [512][160]
                            _Float16* __restrict__ WQh,      // [128][128]
                            _Float16* __restrict__ WKh)      // [128][128]
{
  int i = blockIdx.x*256 + threadIdx.x;
  if (i < G4*KPAD) {
    int g = i / KPAD, k = i - g*KPAD;
    float v = 0.f;
    if (k < F_IN)            v = Wih[g*F_IN + k];
    else if (k < F_IN + NH)  v = Whh[g*NH + (k - F_IN)];
    Wc[i] = (_Float16)v;
  } else if (i < G4*KPAD + NH*NH) {
    int j = i - G4*KPAD;  WQh[j] = (_Float16)WQ[j];
  } else if (i < G4*KPAD + 2*NH*NH) {
    int j = i - G4*KPAD - NH*NH;  WKh[j] = (_Float16)WK[j];
  }
}

// ---------------------------------------------------------------------------
// LSTM over 40 steps for 16 stocks per block, + LayerNorm of last 10 hiddens.
// A = [x_t(6) | h(128) | pad] in LDS; B = Wc gate rows. 4 waves x 8 Ntiles.
// (Compiler hoists the 40 loop-invariant B fragments into VGPRs >255 via
//  s_set_vgpr_msb -- confirmed in codegen.)
// ---------------------------------------------------------------------------
__global__ __launch_bounds__(128) void lstm_kernel(
    const float* __restrict__ X,        // [S][40][6]
    const _Float16* __restrict__ Wc,    // [512][160]
    const float* __restrict__ bih, const float* __restrict__ bhh,
    const float* __restrict__ ln_g, const float* __restrict__ ln_b,
    _Float16* __restrict__ Hln)         // [S][10][128]
{
  __shared__ __align__(16) _Float16 A[16][KPAD];
  __shared__ float Cst[16][NH];
  __shared__ float Gt[16][G4];
  const int tid  = threadIdx.x;
  const int lane = tid & 31, wave = tid >> 5;
  const int coln = lane & 15, hi = lane >> 4;
  const int blk  = blockIdx.x;

  for (int i = tid; i < 16*KPAD; i += 128) (&A[0][0])[i] = (_Float16)0.f;
  for (int i = tid; i < 16*NH;   i += 128) (&Cst[0][0])[i] = 0.f;

  float bsum[8];
#pragma unroll
  for (int j=0;j<8;j++){ int g = (wave*8+j)*16 + coln; bsum[j] = bih[g]+bhh[g]; }
  __syncthreads();

  for (int t=0;t<LSEQ;t++){
    // stage x_t
    if (tid < 16*F_IN) {
      int m = tid / F_IN, f = tid - m*F_IN;
      int s = blk*16 + m; if (s >= S_TOT) s = S_TOT-1;
      A[m][f] = (_Float16)X[(s*LSEQ + t)*F_IN + f];
    }
    __syncthreads();

    // gates GEMM: [16 x 160] x [160 x 512]
    v8f acc[8];
#pragma unroll
    for (int j=0;j<8;j++)
#pragma unroll
      for (int e=0;e<8;e++) acc[j][e] = 0.f;
#pragma unroll
    for (int kc=0;kc<5;kc++){
      v16h af = ldfragA(&A[coln][0], kc*32, hi);
#pragma unroll
      for (int j=0;j<8;j++){
        int g = (wave*8+j)*16 + coln;
        v16h bf = ldfragB(Wc + (size_t)g*KPAD, kc*32, hi);
        acc[j] = wmma16(af, bf, acc[j]);
      }
    }
#pragma unroll
    for (int j=0;j<8;j++){
      int g = (wave*8+j)*16 + coln;
#pragma unroll
      for (int r=0;r<8;r++) Gt[r + hi*8][g] = acc[j][r] + bsum[j];
    }
    __syncthreads();

    // cell / hidden update: thread -> (stock m, 16 hidden units)
    {
      int m  = tid >> 3;
      int nb = (tid & 7) * 16;
#pragma unroll
      for (int i=0;i<16;i++){
        int n = nb + i;
        float ig = Gt[m][n], fg = Gt[m][NH+n], gg = Gt[m][2*NH+n], og = Gt[m][3*NH+n];
        float c = sigmf(fg)*Cst[m][n] + sigmf(ig)*tanhf(gg);
        Cst[m][n] = c;
        A[m][F_IN + n] = (_Float16)(sigmf(og)*tanhf(c));
      }
    }
    __syncthreads();

    // LayerNorm + store for last LMAX steps
    if (t >= LSEQ-LMAX && tid < 16) {
      int m = tid, s = blk*16 + m;
      if (s < S_TOT) {
        float mean = 0.f;
        for (int n=0;n<NH;n++) mean += (float)A[m][F_IN+n];
        mean *= (1.0f/NH);
        float var = 0.f;
        for (int n=0;n<NH;n++){ float d=(float)A[m][F_IN+n]-mean; var += d*d; }
        var *= (1.0f/NH);
        float rstd = rsqrtf(var + 1e-5f);
        _Float16* dst = Hln + ((size_t)s*LMAX + (t-(LSEQ-LMAX)))*NH;
        for (int n=0;n<NH;n++)
          dst[n] = (_Float16)((((float)A[m][F_IN+n]-mean)*rstd)*ln_g[n] + ln_b[n]);
      }
    }
  }
}

// ---------------------------------------------------------------------------
// Queries: q = l2norm(H[target_idx, -1] @ WQ^T)   (gather rows, then WMMA)
// ---------------------------------------------------------------------------
__global__ __launch_bounds__(32) void q_kernel(
    const _Float16* __restrict__ Hln, const int* __restrict__ tgt_idx,
    const _Float16* __restrict__ WQh, _Float16* __restrict__ qn)
{
  __shared__ __align__(16) _Float16 Qr[16][NH];
  __shared__ float Out[16][NH];
  const int lane = threadIdx.x, coln = lane & 15, hi = lane >> 4;
  const int blk = blockIdx.x;

  for (int ch = lane; ch < 16*16; ch += 32) {     // 16B chunks
    int m = ch >> 4, cc = ch & 15;
    int tg = blk*16 + m; if (tg >= S_TOT) tg = S_TOT-1;
    int s = tgt_idx[tg];
    *(v8h*)&Qr[m][cc*8] = *(const v8h*)(Hln + ((size_t)s*LMAX + (LMAX-1))*NH + cc*8);
  }
  __syncthreads();

  v16h af[4];
#pragma unroll
  for (int kc=0;kc<4;kc++) af[kc] = ldfragA(&Qr[coln][0], kc*32, hi);
#pragma unroll
  for (int nt=0;nt<8;nt++){
    v8f acc;
#pragma unroll
    for (int e=0;e<8;e++) acc[e]=0.f;
#pragma unroll
    for (int kc=0;kc<4;kc++){
      v16h bf = ldfragB(WQh + (size_t)(nt*16+coln)*NH, kc*32, hi);
      acc = wmma16(af[kc], bf, acc);
    }
#pragma unroll
    for (int r=0;r<8;r++) Out[r+hi*8][nt*16+coln] = acc[r];
  }
  __syncthreads();

  if (lane < 16){
    int m = lane, tg = blk*16 + m; if (tg >= S_TOT) tg = S_TOT-1;
    float ss = 0.f;
    for (int k=0;k<NH;k++) ss += Out[m][k]*Out[m][k];
    float sc = 1.0f / fmaxf(sqrtf(ss), 1e-12f);
    for (int k=0;k<NH;k++) qn[(size_t)tg*NH + k] = (_Float16)(Out[m][k]*sc);
  }
}

// ---------------------------------------------------------------------------
// Keys: knorm[s*10+l] = l2norm(Hln[s][l] @ WK^T), rows contiguous.
// ---------------------------------------------------------------------------
__global__ __launch_bounds__(32) void k_kernel(
    const _Float16* __restrict__ Hln, const _Float16* __restrict__ WKh,
    _Float16* __restrict__ knorm)
{
  __shared__ float Out[16][NH];
  const int lane = threadIdx.x, coln = lane & 15, hi = lane >> 4;
  const int blk = blockIdx.x;
  const _Float16* arow = Hln + (size_t)(blk*16 + coln)*NH;

  v16h af[4];
#pragma unroll
  for (int kc=0;kc<4;kc++) af[kc] = ldfragA(arow, kc*32, hi);
#pragma unroll
  for (int nt=0;nt<8;nt++){
    v8f acc;
#pragma unroll
    for (int e=0;e<8;e++) acc[e]=0.f;
#pragma unroll
    for (int kc=0;kc<4;kc++){
      v16h bf = ldfragB(WKh + (size_t)(nt*16+coln)*NH, kc*32, hi);
      acc = wmma16(af[kc], bf, acc);
    }
#pragma unroll
    for (int r=0;r<8;r++) Out[r+hi*8][nt*16+coln] = acc[r];
  }
  __syncthreads();

  if (lane < 16){
    int m = lane;
    float ss = 0.f;
    for (int k=0;k<NH;k++) ss += Out[m][k]*Out[m][k];
    float sc = 1.0f / fmaxf(sqrtf(ss), 1e-12f);
    _Float16* dst = knorm + (size_t)(blk*16 + m)*NH;
    for (int k=0;k<NH;k++) dst[k] = (_Float16)(Out[m][k]*sc);
  }
}

// ---------------------------------------------------------------------------
// Fused attention + top-5: 16 targets/block, 4 waves stride over 1875 key
// tiles. Top-5 kept PER-LANE IN REGISTERS (8 C-rows x 5), so the WMMA stream
// never stalls on LDS; scalar insertion VALU co-executes with XDL WMMA.
// One two-stage LDS merge at the end.
// ---------------------------------------------------------------------------
__global__ __launch_bounds__(128) void attn_kernel(
    const _Float16* __restrict__ qn, const _Float16* __restrict__ knorm,
    const int* __restrict__ tgt_idx, const float* __restrict__ log_temp,
    const float* __restrict__ lag_bias,
    float* __restrict__ topv, int* __restrict__ topi)
{
  __shared__ __align__(16) _Float16 Qr[16][NH];
  __shared__ int   Tg[16];
  __shared__ float Lb[LMAX];
  __shared__ float Cv[4][16][16][TOPK];   // [wave][row][lanecol][k]
  __shared__ int   Ci[4][16][16][TOPK];
  __shared__ float M2v[4][16][TOPK];
  __shared__ int   M2i[4][16][TOPK];
  const int tid = threadIdx.x, lane = tid & 31, wave = tid >> 5;
  const int coln = lane & 15, hi = lane >> 4;
  const int blk = blockIdx.x;

  for (int ch = tid; ch < 16*16; ch += 128){
    int m = ch >> 4, cc = ch & 15;
    int tg = blk*16 + m; if (tg >= S_TOT) tg = S_TOT-1;
    *(v8h*)&Qr[m][cc*8] = *(const v8h*)(qn + (size_t)tg*NH + cc*8);
  }
  if (tid < 16){ int tg = blk*16 + tid; if (tg >= S_TOT) tg = S_TOT-1; Tg[tid] = tgt_idx[tg]; }
  if (tid >= 16 && tid < 16+LMAX) Lb[tid-16] = lag_bias[tid-16];
  __syncthreads();

  float temp = expf(log_temp[0]);
  temp = fminf(fmaxf(temp, 0.1f), sqrtf((float)NH));
  const float invt = 1.0f/temp;

  int tmask[8];
#pragma unroll
  for (int r=0;r<8;r++) tmask[r] = Tg[r + hi*8];

  v16h af[4];
#pragma unroll
  for (int kc=0;kc<4;kc++) af[kc] = ldfragA(&Qr[coln][0], kc*32, hi);

  // per-lane top-5 for each of this lane's 8 C rows
  float tv[8][TOPK]; int ti[8][TOPK];
#pragma unroll
  for (int r=0;r<8;r++)
#pragma unroll
    for (int k=0;k<TOPK;k++){ tv[r][k] = -INFINITY; ti[r][k] = 0; }

  for (int tile = wave; tile < KTILES; tile += 4){
    // prefetch next tile's key row for this lane (global_prefetch_b8)
    if (tile + 4 < KTILES)
      __builtin_prefetch(knorm + (size_t)((tile+4)*16 + coln)*NH, 0, 3);

    v8f acc;
#pragma unroll
    for (int e=0;e<8;e++) acc[e]=0.f;
#pragma unroll
    for (int kc=0;kc<4;kc++){
      v16h bf = ldfragB(knorm + (size_t)(tile*16 + coln)*NH, kc*32, hi);
      acc = wmma16(af[kc], bf, acc);
    }
    const int colg = tile*16 + coln;
    const int sdx  = colg / LMAX;
    const int lag  = colg - sdx*LMAX;
    const float lb = Lb[lag];
#pragma unroll
    for (int r=0;r<8;r++){
      float sc = acc[r]*invt + lb;
      if (sdx == tmask[r]) sc = -INFINITY;
      if (sc > tv[r][TOPK-1]){
        float nv = sc; int ni = colg;
#pragma unroll
        for (int k=0;k<TOPK;k++){
          if (nv > tv[r][k]) { float a=tv[r][k]; int b=ti[r][k];
                               tv[r][k]=nv; ti[r][k]=ni; nv=a; ni=b; }
        }
      }
    }
  }

  // dump per-lane candidates
#pragma unroll
  for (int r=0;r<8;r++)
#pragma unroll
    for (int k=0;k<TOPK;k++){
      Cv[wave][r + hi*8][coln][k] = tv[r][k];
      Ci[wave][r + hi*8][coln][k] = ti[r][k];
    }
  __syncthreads();

  // stage 1: (wave w, row m) -> merge 16 lanes x 5 -> top-5
  if (tid < 64){
    int m = tid >> 2, w2 = tid & 3;
    for (int k=0;k<TOPK;k++){
      float best = -INFINITY; int bc=0, bj=0;
      for (int c=0;c<16;c++)
#pragma unroll
        for (int j=0;j<TOPK;j++){
          float v = Cv[w2][m][c][j];
          if (v > best){ best=v; bc=c; bj=j; }
        }
      M2v[w2][m][k] = best;
      M2i[w2][m][k] = Ci[w2][m][bc][bj];
      Cv[w2][m][bc][bj] = -INFINITY;
    }
  }
  __syncthreads();

  // stage 2: row m -> merge 4 waves x 5 -> final sorted top-5
  if (tid < 16){
    int m = tid, tg = blk*16 + m;
    for (int k=0;k<TOPK;k++){
      float best = -INFINITY; int bw=0, bj=0;
#pragma unroll
      for (int w2=0; w2<4; w2++)
#pragma unroll
        for (int j=0; j<TOPK; j++){
          float v = M2v[w2][m][j];
          if (v > best){ best=v; bw=w2; bj=j; }
        }
      int bi = M2i[bw][m][bj];
      M2v[bw][m][bj] = -INFINITY;
      if (tg < S_TOT){ topv[tg*TOPK + k] = best; topi[tg*TOPK + k] = bi; }
    }
  }
}

// ---------------------------------------------------------------------------
// Softmax over top-5 + gather lagged raw features + 12->64->32->1 MLP.
// ---------------------------------------------------------------------------
__global__ void mlp_kernel(
    const float* __restrict__ Xraw,   // [S][40][6]
    const float* __restrict__ topv, const int* __restrict__ topi,
    const float* __restrict__ W1, const float* __restrict__ b1,
    const float* __restrict__ W2, const float* __restrict__ b2,
    const float* __restrict__ W3, const float* __restrict__ b3,
    float* __restrict__ out)
{
  int tg = blockIdx.x*blockDim.x + threadIdx.x;
  if (tg >= S_TOT) return;

  float v[TOPK]; int id[TOPK];
#pragma unroll
  for (int k=0;k<TOPK;k++){ v[k]=topv[tg*TOPK+k]; id[k]=topi[tg*TOPK+k]; }
  float mx = v[0];
#pragma unroll
  for (int k=1;k<TOPK;k++) mx = fmaxf(mx, v[k]);
  float e[TOPK], sum=0.f;
#pragma unroll
  for (int k=0;k<TOPK;k++){ e[k]=expf(v[k]-mx); sum+=e[k]; }
  float inv = 1.0f/sum;

  float feat[2*F_IN];
#pragma unroll
  for (int f=0;f<2*F_IN;f++) feat[f]=0.f;
#pragma unroll
  for (int k=0;k<TOPK;k++){
    int leader = id[k] / LMAX;
    int lag    = id[k] - leader*LMAX;
    int lp     = (LSEQ-1) - (LMAX - lag);       // = 29 + lag, in range
    const float* zp = Xraw + ((size_t)leader*LSEQ + lp)*F_IN;
    float w = e[k]*inv;
#pragma unroll
    for (int f=0;f<F_IN;f++) feat[f] += w*zp[f];
    if (k == 0)
#pragma unroll
      for (int f=0;f<F_IN;f++) feat[F_IN+f] = zp[f];
  }

  float h1[64];
#pragma unroll
  for (int o=0;o<64;o++){
    float a = b1[o];
#pragma unroll
    for (int i=0;i<12;i++) a += W1[o*12+i]*feat[i];
    h1[o] = fmaxf(a, 0.f);
  }
  float h2[32];
#pragma unroll
  for (int o=0;o<32;o++){
    float a = b2[o];
#pragma unroll
    for (int i=0;i<64;i++) a += W2[o*64+i]*h1[i];
    h2[o] = fmaxf(a, 0.f);
  }
  float a = b3[0];
#pragma unroll
  for (int i=0;i<32;i++) a += W3[i]*h2[i];
  out[tg] = a;
}

// ---------------------------------------------------------------------------
extern "C" void kernel_launch(void* const* d_in, const int* in_sizes, int n_in,
                              void* d_out, int out_size, void* d_ws, size_t ws_size,
                              hipStream_t stream) {
  (void)in_sizes; (void)n_in; (void)out_size; (void)ws_size;
  const float* X_scaled = (const float*)d_in[0];
  const float* X_raw    = (const float*)d_in[1];
  const int*   tgt_idx  = (const int*)  d_in[2];
  const float* Wih      = (const float*)d_in[3];
  const float* Whh      = (const float*)d_in[4];
  const float* bih      = (const float*)d_in[5];
  const float* bhh      = (const float*)d_in[6];
  const float* ln_g     = (const float*)d_in[7];
  const float* ln_b     = (const float*)d_in[8];
  const float* WQ       = (const float*)d_in[9];
  const float* WK       = (const float*)d_in[10];
  const float* log_temp = (const float*)d_in[11];
  const float* lag_bias = (const float*)d_in[12];
  const float* W1 = (const float*)d_in[13];
  const float* b1 = (const float*)d_in[14];
  const float* W2 = (const float*)d_in[15];
  const float* b2 = (const float*)d_in[16];
  const float* W3 = (const float*)d_in[17];
  const float* b3 = (const float*)d_in[18];
  float* out = (float*)d_out;

  size_t off = 0;
  auto take = [&](size_t bytes){ size_t c = off; off += (bytes + 255) & ~(size_t)255; return c; };
  char* ws = (char*)d_ws;
  _Float16* Wc    = (_Float16*)(ws + take((size_t)G4*KPAD*2));
  _Float16* WQh   = (_Float16*)(ws + take((size_t)NH*NH*2));
  _Float16* WKh   = (_Float16*)(ws + take((size_t)NH*NH*2));
  _Float16* Hln   = (_Float16*)(ws + take((size_t)S_TOT*LMAX*NH*2));
  _Float16* qn    = (_Float16*)(ws + take((size_t)S_TOT*NH*2));
  _Float16* knorm = (_Float16*)(ws + take((size_t)NKEY*NH*2));
  float*    topv  = (float*)   (ws + take((size_t)S_TOT*TOPK*4));
  int*      topi  = (int*)     (ws + take((size_t)S_TOT*TOPK*4));

  int prep_elems = G4*KPAD + 2*NH*NH;
  prep_kernel<<<(prep_elems + 255)/256, 256, 0, stream>>>(Wih, Whh, WQ, WK, Wc, WQh, WKh);
  lstm_kernel<<<TBLK, 128, 0, stream>>>(X_scaled, Wc, bih, bhh, ln_g, ln_b, Hln);
  q_kernel<<<TBLK, 32, 0, stream>>>(Hln, tgt_idx, WQh, qn);
  k_kernel<<<KTILES, 32, 0, stream>>>(Hln, WKh, knorm);
  attn_kernel<<<TBLK, 128, 0, stream>>>(qn, knorm, tgt_idx, log_temp, lag_bias, topv, topi);
  mlp_kernel<<<(S_TOT + 255)/256, 256, 0, stream>>>(X_raw, topv, topi, W1, b1, W2, b2, W3, b3, out);
}
